// VectorQuantizer_32246614459214
// MI455X (gfx1250) — matrix operand order repaired and checked by reading, not compiled, not run on hardware
//
#include <hip/hip_runtime.h>

// ---------------------------------------------------------------------------
// VQ-VAE vector quantizer for MI455X (gfx1250).
// z: [16, 256, 32, 32] f32, emb: [16384, 256] f32.
// Outputs concatenated in d_out: quant (4194304 f32), loss (1 f32),
// indices (16384, written as f32).
//
// Core: argmin_j (||e_j||^2 - 2 z_i . e_j) as a 16384x16384x256 bf16 WMMA
// GEMM with fused argmin epilogue (distance matrix never materialized).
// Round 2: 32 codes per iteration with TWO independent WMMA accumulator
// chains interleaved (breaks the single-accumulator RAW chain seen in the
// round-1 assembly), 1 barrier per 32 codes.
// ---------------------------------------------------------------------------

typedef __attribute__((ext_vector_type(16))) __bf16 v16bf;
typedef __attribute__((ext_vector_type(8)))  float  v8f;

#define NE        16384   // number of codes
#define DDIM      256     // embedding dim
#define NROWS     16384   // flattened z rows (16*32*32)
#define ROWS_WG   128     // rows per workgroup (8 waves x 16 rows)
#define BLKC      32      // codes per main-loop iteration (2 WMMA column tiles)
#define NBLK      (NE / BLKC)   // 512
#define BSTRIDE   528     // LDS byte stride per code row (512 + 16 pad, 16B aligned)

union Frag { v16bf v; uint4 q[2]; };

struct BBuf { unsigned char B[BLKC * BSTRIDE]; float en[BLKC]; };  // 17024 bytes
struct RedT { float v[8][32][8]; int i[8][32][8]; };               // 16384 bytes

__device__ __forceinline__ unsigned short f2bf(float f) {
    unsigned u = __float_as_uint(f);
    unsigned r = u + 0x7FFFu + ((u >> 16) & 1u);   // round-to-nearest-even
    return (unsigned short)(r >> 16);
}

// ---------------- kernel 1a: emb -> bf16, enorm[j] = ||e_j||^2 --------------
__global__ __launch_bounds__(256)
void vq_emb_conv(const float* __restrict__ emb,
                 unsigned short* __restrict__ emb16,
                 float* __restrict__ enorm) {
    const int j = blockIdx.x;
    const int c = threadIdx.x;
    float v = emb[(size_t)j * DDIM + c];
    emb16[(size_t)j * DDIM + c] = f2bf(v);
    float p = v * v;
    #pragma unroll
    for (int off = 16; off > 0; off >>= 1) p += __shfl_down(p, off, 32);
    __shared__ float ps[8];
    if ((threadIdx.x & 31) == 0) ps[threadIdx.x >> 5] = p;
    __syncthreads();
    if (threadIdx.x == 0) {
        float s = 0.f;
        #pragma unroll
        for (int i = 0; i < 8; ++i) s += ps[i];
        enorm[j] = s;
    }
}

// ---------------- kernel 1b: z [B,C,HW] -> zf16 [N,d] (transpose+bf16) ------
__global__ __launch_bounds__(256)
void vq_z_transpose(const float* __restrict__ z,
                    unsigned short* __restrict__ zf16) {
    __shared__ float tile[32][33];
    const int tx = threadIdx.x, ty = threadIdx.y;
    const int hw0 = blockIdx.x * 32, c0 = blockIdx.y * 32, b = blockIdx.z;
    #pragma unroll
    for (int i = 0; i < 4; ++i) {
        int c = c0 + ty + i * 8;
        tile[ty + i * 8][tx] =
            z[(size_t)b * 262144 + (size_t)c * 1024 + hw0 + tx];  // coalesced in hw
    }
    __syncthreads();
    #pragma unroll
    for (int i = 0; i < 4; ++i) {
        int hwl = ty + i * 8;
        int n = b * 1024 + hw0 + hwl;
        zf16[(size_t)n * DDIM + c0 + tx] = f2bf(tile[tx][hwl]);   // coalesced in c
    }
}

// ---------------- kernel 2: fused bf16 WMMA GEMM + argmin -------------------
// One WG = 8 waves = 128 rows. Each wave owns a 16-row A tile (all 256 dims
// live in 64 VGPRs, ISA 16-bit A layout). All waves sweep all 16384 codes in
// blocks of 32 (two 16-code column tiles -> two independent WMMA accumulator
// chains); B tiles double-buffered in LDS and shared by the 8 waves.
__global__ __launch_bounds__(256)
void vq_argmin(const unsigned short* __restrict__ zf16,
               const unsigned short* __restrict__ emb16,
               const float* __restrict__ enorm,
               int* __restrict__ idx_out,
               float* __restrict__ idxf_out) {
    __shared__ union { BBuf buf[2]; RedT red; } sm;

    const int tid  = threadIdx.x;
    const int lane = tid & 31;
    const int wave = tid >> 5;
    const int h    = lane >> 4;    // lane half: selects K sub-bank per ISA layout
    const int lm   = lane & 15;
    const int n0   = blockIdx.x * ROWS_WG;

    // ---- A fragments: row M = lm of this wave's tile, ISA 16-bit A layout:
    // half 0: K = 32q + {0..7, 16..23}; half 1: K = 32q + {8..15, 24..31}.
    Frag a[8];
    {
        const char* base = (const char*)(zf16 + (size_t)(n0 + wave * 16 + lm) * DDIM);
        #pragma unroll
        for (int q = 0; q < 8; ++q) {
            a[q].q[0] = *(const uint4*)(base + q * 64 + h * 16);
            a[q].q[1] = *(const uint4*)(base + q * 64 + 32 + h * 16);
        }
    }

    // ---- cooperative B staging: 32 codes x 512B, coalesced uint4 copies
    const uint4* embq = (const uint4*)emb16;      // 32 uint4 per code row
    const int srow = tid >> 5;                    // 0..7 (+8/+16/+24 below)
    const int scol = tid & 31;                    // 0..31

    uint4 r0 = embq[(size_t)(srow +  0) * 32 + scol];
    uint4 r1 = embq[(size_t)(srow +  8) * 32 + scol];
    uint4 r2 = embq[(size_t)(srow + 16) * 32 + scol];
    uint4 r3 = embq[(size_t)(srow + 24) * 32 + scol];
    float re = (tid < BLKC) ? enorm[tid] : 0.f;
    *(uint4*)(sm.buf[0].B + (srow +  0) * BSTRIDE + scol * 16) = r0;
    *(uint4*)(sm.buf[0].B + (srow +  8) * BSTRIDE + scol * 16) = r1;
    *(uint4*)(sm.buf[0].B + (srow + 16) * BSTRIDE + scol * 16) = r2;
    *(uint4*)(sm.buf[0].B + (srow + 24) * BSTRIDE + scol * 16) = r3;
    if (tid < BLKC) sm.buf[0].en[tid] = re;
    __syncthreads();

    float minv[8];
    int   mini[8];
    #pragma unroll
    for (int r = 0; r < 8; ++r) { minv[r] = 3.4e38f; mini[r] = 0; }

    int cur = 0;
    #pragma unroll 1
    for (int blk = 0; blk < NBLK; ++blk) {
        const int jbase = blk * BLKC;
        const bool more = (blk + 1 < NBLK);
        if (more) {                               // prefetch next tile to regs
            const size_t jb = (size_t)jbase + BLKC;
            r0 = embq[(jb + srow +  0) * 32 + scol];
            r1 = embq[(jb + srow +  8) * 32 + scol];
            r2 = embq[(jb + srow + 16) * 32 + scol];
            r3 = embq[(jb + srow + 24) * 32 + scol];
            if (tid < BLKC) re = enorm[jb + tid];
        }

        // B fragments from LDS, ISA 16-bit B layout for 32x16:
        // lane N = lm (code), half 0 -> K = 32q + {0..7, 16..23}, half 1 -> K = 32q + {8..15, 24..31}
        // Two column tiles: codes jbase+lm (Bb0) and jbase+16+lm (Bb1).
        const unsigned char* Bb0 = sm.buf[cur].B + lm * BSTRIDE + h * 16;
        const unsigned char* Bb1 = Bb0 + 16 * BSTRIDE;
        v8f acc0 = {};
        v8f acc1 = {};
        #pragma unroll
        for (int q = 0; q < 8; ++q) {
            Frag b0, b1;
            b0.q[0] = *(const uint4*)(Bb0 + q * 64);
            b0.q[1] = *(const uint4*)(Bb0 + q * 64 + 32);
            b1.q[0] = *(const uint4*)(Bb1 + q * 64);
            b1.q[1] = *(const uint4*)(Bb1 + q * 64 + 32);
            acc0 = __builtin_amdgcn_wmma_f32_16x16x32_bf16(
                false, a[q].v, false, b0.v, (short)0, acc0, false, false);
            acc1 = __builtin_amdgcn_wmma_f32_16x16x32_bf16(
                false, a[q].v, false, b1.v, (short)0, acc1, false, false);
        }

        // epilogue: dist proxy = ||e_j||^2 - 2 * (z . e_j); running per-lane min.
        // C layout: lane l, vgpr r -> row M = r + 8*(l>>4), code j = jbase + (l&15)
        const float enj0 = sm.buf[cur].en[lm];
        const float enj1 = sm.buf[cur].en[lm + 16];
        #pragma unroll
        for (int r = 0; r < 8; ++r) {
            float s0 = __builtin_fmaf(-2.0f, acc0[r], enj0);
            float s1 = __builtin_fmaf(-2.0f, acc1[r], enj1);
            if (s0 < minv[r]) { minv[r] = s0; mini[r] = jbase; }
            if (s1 < minv[r]) { minv[r] = s1; mini[r] = jbase + 16; }
        }

        if (more) {
            *(uint4*)(sm.buf[cur ^ 1].B + (srow +  0) * BSTRIDE + scol * 16) = r0;
            *(uint4*)(sm.buf[cur ^ 1].B + (srow +  8) * BSTRIDE + scol * 16) = r1;
            *(uint4*)(sm.buf[cur ^ 1].B + (srow + 16) * BSTRIDE + scol * 16) = r2;
            *(uint4*)(sm.buf[cur ^ 1].B + (srow + 24) * BSTRIDE + scol * 16) = r3;
            if (tid < BLKC) sm.buf[cur ^ 1].en[tid] = re;
        }
        __syncthreads();
        cur ^= 1;
    }

    // ---- final argmin reduction across the 16 lanes of each half-wave ------
    #pragma unroll
    for (int r = 0; r < 8; ++r) {
        sm.red.v[wave][lane][r] = minv[r];
        sm.red.i[wave][lane][r] = mini[r];
    }
    __syncthreads();
    if (tid < 128) {                     // one thread per row of this WG
        const int w = tid >> 4, loc = tid & 15, r = loc & 7, hh = loc >> 3;
        float bv = 3.4e38f; int bi = 0;
        #pragma unroll 1
        for (int k = 0; k < 16; ++k) {   // lane k holds codes with j % 16 == k
            float v = sm.red.v[w][hh * 16 + k][r];
            int  ix = sm.red.i[w][hh * 16 + k][r] + k;
            if (v < bv) { bv = v; bi = ix; }
        }
        const int n = n0 + tid;
        idx_out[n] = bi;
        idxf_out[n] = (float)bi;
    }
}

// ---------------- kernel 3: gather quant + per-block loss partials ----------
__global__ __launch_bounds__(256)
void vq_gather(const float* __restrict__ z, const float* __restrict__ emb,
               const int* __restrict__ idx, float* __restrict__ quant,
               float* __restrict__ partial) {
    const int tx = threadIdx.x & 31, ty = threadIdx.x >> 5;
    const int hw0 = blockIdx.x * 32, c0 = blockIdx.y * 32, b = blockIdx.z;
    const int n = b * 1024 + hw0 + tx;
    const int j = idx[n];
    float acc = 0.f;
    #pragma unroll
    for (int i = 0; i < 4; ++i) {
        int c = c0 + ty + i * 8;
        float e = emb[(size_t)j * DDIM + c];
        size_t off = (size_t)b * 262144 + (size_t)c * 1024 + hw0 + tx;
        float zv = z[off];
        quant[off] = e;                       // straight-through: quant == z_q
        float d = e - zv;
        acc = __builtin_fmaf(d, d, acc);
    }
    #pragma unroll
    for (int off = 16; off > 0; off >>= 1) acc += __shfl_down(acc, off, 32);
    __shared__ float ps[8];
    if (tx == 0) ps[ty] = acc;
    __syncthreads();
    if (threadIdx.x == 0) {
        float s = 0.f;
        #pragma unroll
        for (int i = 0; i < 8; ++i) s += ps[i];
        partial[blockIdx.z * 256 + blockIdx.y * 32 + blockIdx.x] = s;
    }
}

// ---------------- kernel 4: deterministic fixed-order loss reduction --------
__global__ __launch_bounds__(256)
void vq_loss_final(const float* __restrict__ partial, float* __restrict__ loss) {
    float s = 0.f;
    #pragma unroll 1
    for (int i = 0; i < 16; ++i) s += partial[threadIdx.x + i * 256];
    #pragma unroll
    for (int off = 16; off > 0; off >>= 1) s += __shfl_down(s, off, 32);
    __shared__ float ps[8];
    if ((threadIdx.x & 31) == 0) ps[threadIdx.x >> 5] = s;
    __syncthreads();
    if (threadIdx.x == 0) {
        float t = 0.f;
        #pragma unroll
        for (int i = 0; i < 8; ++i) t += ps[i];
        // loss = (1 + beta) * mean((z_q - z)^2), beta = 0.25
        *loss = t * (1.25f / ((float)NROWS * (float)DDIM));
    }
}

// ---------------------------------------------------------------------------
extern "C" void kernel_launch(void* const* d_in, const int* in_sizes, int n_in,
                              void* d_out, int out_size, void* d_ws, size_t ws_size,
                              hipStream_t stream) {
    const float* z   = (const float*)d_in[0];   // 16*256*32*32
    const float* emb = (const float*)d_in[1];   // 16384*256

    float* out   = (float*)d_out;
    float* quant = out;                         // 4194304 f32
    float* loss  = out + 4194304;               // 1 f32
    float* idxf  = out + 4194305;               // 16384 f32

    char* ws = (char*)d_ws;
    unsigned short* emb16 = (unsigned short*)(ws);              // 8 MB bf16
    unsigned short* zf16  = (unsigned short*)(ws + 8388608);    // 8 MB bf16
    float* enorm          = (float*)(ws + 16777216);            // 64 KB
    int*   idx            = (int*)(ws + 16842752);              // 64 KB
    float* partial        = (float*)(ws + 16908288);            // 16 KB

    vq_emb_conv  <<<NE, 256, 0, stream>>>(emb, emb16, enorm);
    vq_z_transpose<<<dim3(32, 8, 16), dim3(32, 8), 0, stream>>>(z, zf16);
    vq_argmin    <<<NROWS / ROWS_WG, 256, 0, stream>>>(zf16, emb16, enorm, idx, idxf);
    vq_gather    <<<dim3(32, 8, 16), 256, 0, stream>>>(z, emb, idx, quant, partial);
    vq_loss_final<<<1, 256, 0, stream>>>(partial, loss);
}
